// TDSLSTMModel_10170482557417
// MI455X (gfx1250) — compile-verified, matrix operand order checked by reading, not run on hardware
//
#include <hip/hip_runtime.h>
#include <hip/hip_bf16.h>
#include <math.h>

// ---------------------------------------------------------------------------
// Types for CDNA5 WMMA (wave32): D(16x16,f32) = A(16x32,f16) x B(32x16,f16) + C
// ---------------------------------------------------------------------------
typedef __attribute__((ext_vector_type(16))) _Float16 v16h;
typedef __attribute__((ext_vector_type(8)))  float    v8f;

__device__ __forceinline__ v8f wmma_f16(v16h a, v16h b, v8f c) {
  return __builtin_amdgcn_wmma_f32_16x16x32_f16(
      /*neg_a=*/false, a, /*neg_b=*/false, b,
      /*c_mod=*/(short)0, c, /*reuse_a=*/false, /*reuse_b=*/false);
}

// Load one 16x32 f16 tile (A layout; B assumed symmetric with lane = N).
// Row r = lane&15, per-lane halves: k = 16*(i/8) + 8*(lane>>4) + (i%8).
__device__ __forceinline__ v16h load_tile16x32(const _Float16* __restrict__ base,
                                               int stride, int lane) {
  const _Float16* r = base + (size_t)(lane & 15) * stride + ((lane >> 4) << 3);
  v16h a;
#pragma unroll
  for (int g = 0; g < 2; ++g)
#pragma unroll
    for (int j = 0; j < 8; ++j)
      a[g * 8 + j] = r[g * 16 + j];
  return a;
}

// ---------------------------------------------------------------------------
// CDNA5 async global->LDS copies (ASYNCcnt-tracked), per CDNA5 ISA.
// LDS operand is the wave-relative LDS byte address == low 32 bits of the
// generic pointer (ISA 10.2: LDS aperture uses addr[31:0] directly).
// ---------------------------------------------------------------------------
__device__ __forceinline__ void async_ld_b128(void* lds, const void* gaddr) {
  unsigned off = (unsigned)(size_t)lds;
  asm volatile("global_load_async_to_lds_b128 %0, %1, off"
               :: "v"(off), "v"(gaddr) : "memory");
}
__device__ __forceinline__ void async_ld_b32(void* lds, const void* gaddr) {
  unsigned off = (unsigned)(size_t)lds;
  asm volatile("global_load_async_to_lds_b32 %0, %1, off"
               :: "v"(off), "v"(gaddr) : "memory");
}
__device__ __forceinline__ void wait_async0() {
  asm volatile("s_wait_asynccnt 0x0" ::: "memory");
}

__device__ __forceinline__ float sigmoidf_(float x) {
  return 1.f / (1.f + __expf(-x));
}

// ---------------------------------------------------------------------------
// Model constants
// ---------------------------------------------------------------------------
#define BB   64
#define TT0  16000
#define TT1  7998
#define TT2  3997
#define TTt  99
#define DOUT 20
#define HH   128

// ---------------------------------------------------------------------------
// Prep kernels: f32 -> f16 weight conversion (with optional padding), bias add
// ---------------------------------------------------------------------------
__global__ void k_f16copy(const float* __restrict__ s, _Float16* __restrict__ d, int n) {
  int i = blockIdx.x * 256 + threadIdx.x;
  if (i < n) d[i] = (_Float16)s[i];
}
__global__ void k_f16padcols(const float* __restrict__ s, _Float16* __restrict__ d,
                             int rows, int sk, int dk) {
  int i = blockIdx.x * 256 + threadIdx.x;
  if (i >= rows * dk) return;
  int r = i / dk, k = i % dk;
  d[i] = (k < sk) ? (_Float16)s[r * sk + k] : (_Float16)0.f;
}
__global__ void k_f16padrows(const float* __restrict__ s, _Float16* __restrict__ d,
                             int srows, int drows, int kk) {
  int i = blockIdx.x * 256 + threadIdx.x;
  if (i >= drows * kk) return;
  int r = i / kk;
  d[i] = (r < srows) ? (_Float16)s[i] : (_Float16)0.f;
}
__global__ void k_vecadd(const float* __restrict__ a, const float* __restrict__ b,
                         float* __restrict__ d, int n) {
  int i = blockIdx.x * 256 + threadIdx.x;
  if (i < n) d[i] = a[i] + b[i];
}

// ---------------------------------------------------------------------------
// Stage input convolutions (VALU fp32; small FLOP share, bandwidth bound)
// out is channel-major (B, Cout, T)
// ---------------------------------------------------------------------------
__global__ __launch_bounds__(256) void k_inconv1(const float* __restrict__ x,
                                                 const float* __restrict__ W,
                                                 const float* __restrict__ bias,
                                                 float* __restrict__ out) {
  int idx = blockIdx.x * 256 + threadIdx.x;            // total = B*T1*32
  int co = idx & 31;
  int rest = idx >> 5;
  int t = rest % TT1;
  int b = rest / TT1;
  float s = bias[co];
  const float* xb = x + ((size_t)b * TT0 + 2 * t) * 16;  // (B,T,Cin) layout
  const float* w  = W + co * 16 * 5;
#pragma unroll
  for (int k = 0; k < 5; ++k)
#pragma unroll
    for (int ci = 0; ci < 16; ++ci)
      s += xb[k * 16 + ci] * w[ci * 5 + k];
  out[((size_t)b * 32 + co) * TT1 + t] = fmaxf(s, 0.f);
}

__global__ __launch_bounds__(256) void k_inconv2(const float* __restrict__ act,
                                                 const float* __restrict__ W,
                                                 const float* __restrict__ bias,
                                                 float* __restrict__ out) {
  int idx = blockIdx.x * 256 + threadIdx.x;            // total = B*T2*64
  int co = idx & 63;
  int rest = idx >> 6;
  int t = rest % TT2;
  int b = rest / TT2;
  float s = bias[co];
  const float* ab = act + (size_t)b * 32 * TT1;
  const float* w  = W + co * 32 * 5;
#pragma unroll
  for (int k = 0; k < 5; ++k)
    for (int ci = 0; ci < 32; ++ci)
      s += ab[(size_t)ci * TT1 + 2 * t + k] * w[ci * 5 + k];
  out[((size_t)b * 64 + co) * TT2 + t] = fmaxf(s, 0.f);
}

// ---------------------------------------------------------------------------
// TDS block part 1: causal time-conv2d + relu + residual + LayerNorm.
// act: channel-major (B, cw, T) fp32.  Input tile + weights staged into LDS
// with async global->LDS copies (ASYNCcnt). Outputs token-major y (f16 for
// GEMM A, f32 for residual). One block = (batch b, 64-token tile).
// ---------------------------------------------------------------------------
__global__ __launch_bounds__(256) void k_tds_conv(const float* __restrict__ act,
                                                  const float* __restrict__ Wc,
                                                  const float* __restrict__ bc,
                                                  const float* __restrict__ g,
                                                  const float* __restrict__ beta,
                                                  _Float16* __restrict__ y16,
                                                  float* __restrict__ y32,
                                                  int Cc, int Ww, int kw, int T) {
  const int cw = Cc * Ww;
  const int nT = (T + 63) >> 6;
  const int b  = blockIdx.x / nT;
  const int t0 = (blockIdx.x % nT) << 6;
  const int win = 64 + kw - 1;
  const int tid = threadIdx.x;

  __shared__ __align__(16) float sIn[64 * 84];     // cw<=64, win<=84
  __shared__ __align__(16) float sOut[64 * 64];    // [tok][ch]
  __shared__ __align__(16) float sW[8 * 8 * 21];

  // async-stage conv weights and the input window into LDS
  for (int i = tid; i < Cc * Cc * kw; i += 256)
    async_ld_b32(&sW[i], &Wc[i]);
  for (int i = tid; i < cw * win; i += 256) {
    int ch = i / win, ttv = i % win;
    int t = t0 + ttv - (kw - 1);
    if (t >= 0 && t < T)
      async_ld_b32(&sIn[i], &act[((size_t)b * cw + ch) * T + t]);
    else
      sIn[i] = 0.f;
  }
  wait_async0();
  __syncthreads();

  for (int e = tid; e < 64 * cw; e += 256) {
    int tt = e / cw, ch = e % cw;
    int t = t0 + tt;
    if (t >= T) continue;
    int c = ch / Ww, w = ch % Ww;
    float s = bc[c];
    for (int ci = 0; ci < Cc; ++ci) {
      const float* wrow = &sW[(c * Cc + ci) * kw];
      const float* irow = &sIn[(ci * Ww + w) * win + tt];
      for (int k = 0; k < kw; ++k) s += wrow[k] * irow[k];
    }
    float r = fmaxf(s, 0.f);
    sOut[tt * cw + ch] = r + sIn[ch * win + tt + kw - 1];  // residual at time t
  }
  __syncthreads();

  if (tid < 64) {
    int tt = tid, t = t0 + tt;
    if (t < T) {
      float m = 0.f;
      for (int ch = 0; ch < cw; ++ch) m += sOut[tt * cw + ch];
      m /= cw;
      float v = 0.f;
      for (int ch = 0; ch < cw; ++ch) {
        float d = sOut[tt * cw + ch] - m;
        v += d * d;
      }
      float inv = rsqrtf(v / cw + 1e-5f);
      size_t tok = (size_t)b * T + t;
      for (int ch = 0; ch < cw; ++ch) {
        float val = (sOut[tt * cw + ch] - m) * inv * g[ch] + beta[ch];
        y32[tok * cw + ch] = val;
        y16[tok * cw + ch] = (_Float16)val;
      }
    }
  }
}

// ---------------------------------------------------------------------------
// TDS FC1: h = relu(y @ W1^T + b1), token-major, f16 WMMA (K = CW).
// CW is a template constant so the N-tile loop fully unrolls and the
// accumulators stay in statically-allocated VGPRs (no movrel indexing).
// Weights (<=8KB f16) async-staged to LDS once per block; one wave per
// M-tile computes ALL N-tiles so each A tile is loaded exactly once/K-chunk.
// ---------------------------------------------------------------------------
template <int CW>
__global__ __launch_bounds__(256) void k_fc1_t(const _Float16* __restrict__ A,
                                               const _Float16* __restrict__ W,
                                               const float* __restrict__ bias,
                                               _Float16* __restrict__ Out,
                                               int Mtiles) {
  constexpr int NT = CW / 16;
  __shared__ __align__(16) _Float16 sWl[CW * CW];
  int tid = threadIdx.x, lane = tid & 31, wave = tid >> 5;
  for (int i = tid * 8; i < CW * CW; i += 256 * 8)
    async_ld_b128(&sWl[i], &W[i]);
  wait_async0();
  __syncthreads();

  int mt = blockIdx.x * 8 + wave;
  if (mt >= Mtiles) return;                     // wave-uniform
  v8f acc[NT];
#pragma unroll
  for (int nt = 0; nt < NT; ++nt) acc[nt] = (v8f){};
  for (int kc = 0; kc < CW; kc += 32) {
    __builtin_prefetch(A + ((size_t)mt * 16 + 16) * CW + kc, 0, 1);
    v16h a = load_tile16x32(A + (size_t)mt * 16 * CW + kc, CW, lane);
#pragma unroll
    for (int nt = 0; nt < NT; ++nt) {
      v16h b = load_tile16x32(sWl + (size_t)nt * 16 * CW + kc, CW, lane);
      acc[nt] = wmma_f16(a, b, acc[nt]);
    }
  }
  int ncol = lane & 15;
  int mbase = mt * 16 + ((lane >> 4) << 3);
#pragma unroll
  for (int nt = 0; nt < NT; ++nt) {
    int n = nt * 16 + ncol;
    float bn = bias[n];
#pragma unroll
    for (int r = 0; r < 8; ++r) {
      float v = acc[nt][r] + bn;
      Out[(size_t)(mbase + r) * CW + n] = (_Float16)fmaxf(v, 0.f);
    }
  }
}

// ---------------------------------------------------------------------------
// TDS FC2 + residual + LayerNorm, fused (CW compile-time). One wave = 16
// tokens x all N-tiles; LN row statistics reduce across the 16 lanes holding
// that row (D layout: lane = column, VGPR r + 8*(lane>=16) = row). Output
// scattered back to channel-major (B, cw, T) fp32 for the next conv.
// ---------------------------------------------------------------------------
template <int CW>
__global__ __launch_bounds__(256) void k_fc2ln_t(const _Float16* __restrict__ A,
                                                 const _Float16* __restrict__ W,
                                                 const float* __restrict__ b2,
                                                 const float* __restrict__ yres,
                                                 const float* __restrict__ g,
                                                 const float* __restrict__ beta,
                                                 float* __restrict__ out,
                                                 int Mtiles, int T) {
  constexpr int NT = CW / 16;
  __shared__ __align__(16) _Float16 sWl[CW * CW];
  int tid = threadIdx.x, lane = tid & 31, wave = tid >> 5;
  for (int i = tid * 8; i < CW * CW; i += 256 * 8)
    async_ld_b128(&sWl[i], &W[i]);
  wait_async0();
  __syncthreads();

  int mt = blockIdx.x * 8 + wave;
  if (mt >= Mtiles) return;                     // wave-uniform
  v8f acc[NT];
#pragma unroll
  for (int nt = 0; nt < NT; ++nt) acc[nt] = (v8f){};
  for (int kc = 0; kc < CW; kc += 32) {
    v16h a = load_tile16x32(A + (size_t)mt * 16 * CW + kc, CW, lane);
#pragma unroll
    for (int nt = 0; nt < NT; ++nt) {
      v16h b = load_tile16x32(sWl + (size_t)nt * 16 * CW + kc, CW, lane);
      acc[nt] = wmma_f16(a, b, acc[nt]);
    }
  }
  int ncol = lane & 15;
  int half8 = (lane >> 4) << 3;
#pragma unroll
  for (int nt = 0; nt < NT; ++nt) {
    int n = nt * 16 + ncol;
    float bb = b2[n];
#pragma unroll
    for (int r = 0; r < 8; ++r) {
      int tok = mt * 16 + half8 + r;
      acc[nt][r] += bb + yres[(size_t)tok * CW + n];
    }
  }
  float mean[8], inv[8];
#pragma unroll
  for (int r = 0; r < 8; ++r) {
    float s = 0.f, q = 0.f;
#pragma unroll
    for (int nt = 0; nt < NT; ++nt) {
      float v = acc[nt][r];
      s += v; q += v * v;
    }
#pragma unroll
    for (int m = 1; m < 16; m <<= 1) {          // reduce within 16-lane half
      s += __shfl_xor(s, m, 32);
      q += __shfl_xor(q, m, 32);
    }
    float mu = s / CW;
    mean[r] = mu;
    inv[r] = rsqrtf(q / CW - mu * mu + 1e-5f);
  }
#pragma unroll
  for (int nt = 0; nt < NT; ++nt) {
    int n = nt * 16 + ncol;
    float gg = g[n], be = beta[n];
#pragma unroll
    for (int r = 0; r < 8; ++r) {
      int tok = mt * 16 + half8 + r;
      int b = tok / T, t = tok % T;
      out[((size_t)b * CW + n) * T + t] = (acc[nt][r] - mean[r]) * inv[r] * gg + be;
    }
  }
}

// ---------------------------------------------------------------------------
// Linear interpolation of encoder features T2 -> Tt, f16 out: [t][b][64]
// ---------------------------------------------------------------------------
__global__ __launch_bounds__(256) void k_interp_feats(const float* __restrict__ act,
                                                      _Float16* __restrict__ ft) {
  int idx = blockIdx.x * 256 + threadIdx.x;   // = (t*64 + b)*64 + k
  int k = idx & 63;
  int rest = idx >> 6;
  int b = rest & 63;
  int t = rest >> 6;
  float pos = t * (3996.f / 98.f);
  int lo = (int)floorf(pos);
  if (lo > TT2 - 2) lo = TT2 - 2;
  float w = pos - (float)lo;
  const float* a = act + ((size_t)b * 64 + k) * TT2;
  ft[idx] = (_Float16)(a[lo] * (1.f - w) + a[lo + 1] * w);
}

// ---------------------------------------------------------------------------
// Persistent 2-layer LSTM decoder. One workgroup (32 waves). Per step:
// wave (mt, nt8) computes gate tiles {nt8, nt8+8, nt8+16, nt8+24} so i/f/g/o
// for its hidden slice live in-register; c-state stays in VGPRs across steps;
// h-state in LDS f16 feeds the next GEMM via ds loads; the per-step feature
// slab is async-copied global->LDS (128B per batch row, 16B aligned both
// sides: ft row stride 128B, sU row stride 192B).
// ---------------------------------------------------------------------------
__global__ __launch_bounds__(1024) void k_lstm(const _Float16* __restrict__ ft,   // [99][64][64]
                                               const _Float16* __restrict__ Wi0,  // [512][96]
                                               const _Float16* __restrict__ Wh0,  // [512][128]
                                               const _Float16* __restrict__ Wi1,  // [512][128]
                                               const _Float16* __restrict__ Wh1,  // [512][128]
                                               const _Float16* __restrict__ Wo,   // [32][128]
                                               const float* __restrict__ bz0,
                                               const float* __restrict__ bz1,
                                               const float* __restrict__ bo,
                                               float* __restrict__ poses) {       // [99][64][20]
  __shared__ __align__(16) _Float16 sU[64 * 96];
  __shared__ __align__(16) _Float16 sH0[64 * 128];
  __shared__ __align__(16) _Float16 sH1[64 * 128];
  __shared__ float sPose[64 * 20];

  int tid = threadIdx.x, lane = tid & 31, wave = tid >> 5;
  int mt = wave >> 3, nt8 = wave & 7;
  int ncol = lane & 15, half8 = (lane >> 4) << 3;

  for (int i = tid; i < 64 * 128; i += 1024) { sH0[i] = (_Float16)0.f; sH1[i] = (_Float16)0.f; }
  for (int i = tid; i < 64 * 20; i += 1024) sPose[i] = 0.f;
  v8f c0 = {}, c1 = {};
  __syncthreads();

  for (int t = 0; t < TTt; ++t) {
    // build u = [feat(64) | pose(20) | pad(12)] as f16
    if (tid < 512) {                             // 64 rows x 8 b128 chunks
      int bb = tid >> 3, chunk = tid & 7;
      async_ld_b128(&sU[bb * 96 + chunk * 8],
                    &ft[((size_t)t * 64 + bb) * 64 + chunk * 8]);
    }
    for (int i = tid; i < 64 * 32; i += 1024) {  // tail: pose + zero pad
      int bb = i >> 5, k = 64 + (i & 31);
      sU[bb * 96 + k] = (k < 84) ? (_Float16)sPose[bb * 20 + (k - 64)]
                                 : (_Float16)0.f;
    }
    wait_async0();
    __syncthreads();

    // ---- layer 0 ----
    {
      v8f acc[4] = {{}, {}, {}, {}};
      for (int kc = 0; kc < 96; kc += 32) {
        v16h a = load_tile16x32(sU + mt * 16 * 96 + kc, 96, lane);
#pragma unroll
        for (int gte = 0; gte < 4; ++gte) {
          v16h b = load_tile16x32(Wi0 + (size_t)((gte * 8 + nt8) * 16) * 96 + kc, 96, lane);
          acc[gte] = wmma_f16(a, b, acc[gte]);
        }
      }
      for (int kc = 0; kc < 128; kc += 32) {
        v16h a = load_tile16x32(sH0 + mt * 16 * 128 + kc, 128, lane);
#pragma unroll
        for (int gte = 0; gte < 4; ++gte) {
          v16h b = load_tile16x32(Wh0 + (size_t)((gte * 8 + nt8) * 16) * 128 + kc, 128, lane);
          acc[gte] = wmma_f16(a, b, acc[gte]);
        }
      }
      int n = nt8 * 16 + ncol;
      float b_i = bz0[n], b_f = bz0[128 + n], b_g = bz0[256 + n], b_o = bz0[384 + n];
      __syncthreads();                 // all reads of sH0 complete
#pragma unroll
      for (int r = 0; r < 8; ++r) {
        float iv = sigmoidf_(acc[0][r] + b_i);
        float fv = sigmoidf_(acc[1][r] + b_f);
        float gv = tanhf(acc[2][r] + b_g);
        float ov = sigmoidf_(acc[3][r] + b_o);
        float c = fv * c0[r] + iv * gv;
        c0[r] = c;
        float h = ov * tanhf(c);
        sH0[(mt * 16 + half8 + r) * 128 + n] = (_Float16)h;
      }
      __syncthreads();
    }

    // ---- layer 1 ----
    {
      v8f acc[4] = {{}, {}, {}, {}};
      for (int kc = 0; kc < 128; kc += 32) {
        v16h a = load_tile16x32(sH0 + mt * 16 * 128 + kc, 128, lane);
#pragma unroll
        for (int gte = 0; gte < 4; ++gte) {
          v16h b = load_tile16x32(Wi1 + (size_t)((gte * 8 + nt8) * 16) * 128 + kc, 128, lane);
          acc[gte] = wmma_f16(a, b, acc[gte]);
        }
      }
      for (int kc = 0; kc < 128; kc += 32) {
        v16h a = load_tile16x32(sH1 + mt * 16 * 128 + kc, 128, lane);
#pragma unroll
        for (int gte = 0; gte < 4; ++gte) {
          v16h b = load_tile16x32(Wh1 + (size_t)((gte * 8 + nt8) * 16) * 128 + kc, 128, lane);
          acc[gte] = wmma_f16(a, b, acc[gte]);
        }
      }
      int n = nt8 * 16 + ncol;
      float b_i = bz1[n], b_f = bz1[128 + n], b_g = bz1[256 + n], b_o = bz1[384 + n];
      __syncthreads();
#pragma unroll
      for (int r = 0; r < 8; ++r) {
        float iv = sigmoidf_(acc[0][r] + b_i);
        float fv = sigmoidf_(acc[1][r] + b_f);
        float gv = tanhf(acc[2][r] + b_g);
        float ov = sigmoidf_(acc[3][r] + b_o);
        float c = fv * c1[r] + iv * gv;
        c1[r] = c;
        float h = ov * tanhf(c);
        sH1[(mt * 16 + half8 + r) * 128 + n] = (_Float16)h;
      }
      __syncthreads();
    }

    // ---- pose = h1 @ Wo^T + bo (waves 0..7) ----
    if (wave < 8) {
      int pmt = wave >> 1, pnt = wave & 1;
      v8f pacc = {};
      for (int kc = 0; kc < 128; kc += 32) {
        v16h a = load_tile16x32(sH1 + pmt * 16 * 128 + kc, 128, lane);
        v16h b = load_tile16x32(Wo + (size_t)(pnt * 16) * 128 + kc, 128, lane);
        pacc = wmma_f16(a, b, pacc);
      }
      int d = pnt * 16 + ncol;
      if (d < 20) {
        float bd = bo[d];
#pragma unroll
        for (int r = 0; r < 8; ++r) {
          int m = pmt * 16 + half8 + r;
          float v = pacc[r] + bd;
          sPose[m * 20 + d] = v;
          poses[((size_t)t * 64 + m) * 20 + d] = v;
        }
      }
    }
    __syncthreads();
  }
}

// ---------------------------------------------------------------------------
// Final interpolation Tt -> T, output (B, T, Dout)
// ---------------------------------------------------------------------------
__global__ __launch_bounds__(256) void k_interp_out(const float* __restrict__ poses,
                                                    float* __restrict__ out) {
  int idx = blockIdx.x * 256 + threadIdx.x;   // = (b*16000 + t)*20 + d
  int d = idx % 20;
  int rest = idx / 20;
  int t = rest % TT0;
  int b = rest / TT0;
  float pos = t * (98.f / 15999.f);
  int lo = (int)floorf(pos);
  if (lo > TTt - 2) lo = TTt - 2;
  float w = pos - (float)lo;
  float v = poses[((size_t)lo * 64 + b) * 20 + d] * (1.f - w) +
            poses[((size_t)(lo + 1) * 64 + b) * 20 + d] * w;
  out[idx] = v;
}

// ---------------------------------------------------------------------------
// Host launcher
// ---------------------------------------------------------------------------
extern "C" void kernel_launch(void* const* d_in, const int* in_sizes, int n_in,
                              void* d_out, int out_size, void* d_ws, size_t ws_size,
                              hipStream_t stream) {
  (void)in_sizes; (void)n_in; (void)out_size; (void)ws_size;
  auto F = [&](int i) { return (const float*)d_in[i]; };

  // input index map (setup_inputs flattening order)
  const int IX_X = 0;
  const int IX_WIN0 = 1, IX_BIN0 = 2;
  const int S0B0 = 3, S0B1 = 13;                // block bases (10 tensors each)
  const int IX_WIN1 = 23, IX_BIN1 = 24;
  const int S1B0 = 25, S1B1 = 35;
  const int IX_WI0 = 45, IX_WH0 = 46, IX_BI0 = 47, IX_BH0 = 48;
  const int IX_WI1 = 49, IX_WH1 = 50, IX_BI1 = 51, IX_BH1 = 52;
  const int IX_WO = 53, IX_BO = 54;
  // within block: +0 Wc, +1 bc, +2 g1, +3 b1ln, +4 W1, +5 b1, +6 W2, +7 b2, +8 g2, +9 b2ln

  char* ws = (char*)d_ws;
  size_t off = 0;
  auto take = [&](size_t bytes) -> char* {
    char* p = ws + off;
    off += (bytes + 255) & ~(size_t)255;
    return p;
  };
  float*    actS1 = (float*)take((size_t)BB * 32 * TT1 * 4);
  float*    actS2 = (float*)take((size_t)BB * 64 * TT2 * 4);
  float*    y32   = (float*)take((size_t)BB * TT1 * 32 * 4);   // stage2 fits too
  _Float16* y16   = (_Float16*)take((size_t)BB * TT1 * 32 * 2);
  _Float16* h16   = (_Float16*)take((size_t)BB * TT1 * 32 * 2);
  _Float16* ft16  = (_Float16*)take((size_t)TTt * 64 * 64 * 2);
  float*    poses = (float*)take((size_t)TTt * 64 * 20 * 4);
  _Float16* w1f[4]; _Float16* w2f[4];
  int blkBase[4] = {S0B0, S0B1, S1B0, S1B1};
  int blkCW[4]   = {32, 32, 64, 64};
  for (int i = 0; i < 4; ++i) {
    w1f[i] = (_Float16*)take((size_t)blkCW[i] * blkCW[i] * 2);
    w2f[i] = (_Float16*)take((size_t)blkCW[i] * blkCW[i] * 2);
  }
  _Float16* Wi0f = (_Float16*)take(512 * 96 * 2);
  _Float16* Wh0f = (_Float16*)take(512 * 128 * 2);
  _Float16* Wi1f = (_Float16*)take(512 * 128 * 2);
  _Float16* Wh1f = (_Float16*)take(512 * 128 * 2);
  _Float16* Wof  = (_Float16*)take(32 * 128 * 2);
  float*    bz0  = (float*)take(512 * 4);
  float*    bz1  = (float*)take(512 * 4);

  // ---- weight prep (f32 -> f16, with padding where needed) ----
  for (int i = 0; i < 4; ++i) {
    int n = blkCW[i] * blkCW[i];
    k_f16copy<<<(n + 255) / 256, 256, 0, stream>>>(F(blkBase[i] + 4), w1f[i], n);
    k_f16copy<<<(n + 255) / 256, 256, 0, stream>>>(F(blkBase[i] + 6), w2f[i], n);
  }
  k_f16padcols<<<(512 * 96 + 255) / 256, 256, 0, stream>>>(F(IX_WI0), Wi0f, 512, 84, 96);
  k_f16copy<<<(512 * 128 + 255) / 256, 256, 0, stream>>>(F(IX_WH0), Wh0f, 512 * 128);
  k_f16copy<<<(512 * 128 + 255) / 256, 256, 0, stream>>>(F(IX_WI1), Wi1f, 512 * 128);
  k_f16copy<<<(512 * 128 + 255) / 256, 256, 0, stream>>>(F(IX_WH1), Wh1f, 512 * 128);
  k_f16padrows<<<(32 * 128 + 255) / 256, 256, 0, stream>>>(F(IX_WO), Wof, 20, 32, 128);
  k_vecadd<<<2, 256, 0, stream>>>(F(IX_BI0), F(IX_BH0), bz0, 512);
  k_vecadd<<<2, 256, 0, stream>>>(F(IX_BI1), F(IX_BH1), bz1, 512);

  // ---- stage 1 ----
  k_inconv1<<<(BB * TT1 * 32) / 256, 256, 0, stream>>>(F(IX_X), F(IX_WIN0), F(IX_BIN0), actS1);
  for (int blk = 0; blk < 2; ++blk) {
    int base = blkBase[blk];
    const int Mtiles = BB * TT1 / 16;         // 31992
    k_tds_conv<<<BB * ((TT1 + 63) / 64), 256, 0, stream>>>(
        actS1, F(base + 0), F(base + 1), F(base + 2), F(base + 3),
        y16, y32, 8, 4, 21, TT1);
    k_fc1_t<32><<<(Mtiles + 7) / 8, 256, 0, stream>>>(y16, w1f[blk], F(base + 5),
                                                      h16, Mtiles);
    k_fc2ln_t<32><<<(Mtiles + 7) / 8, 256, 0, stream>>>(h16, w2f[blk], F(base + 7),
                                                        y32, F(base + 8), F(base + 9),
                                                        actS1, Mtiles, TT1);
  }

  // ---- stage 2 ----
  k_inconv2<<<(BB * TT2 * 64) / 256, 256, 0, stream>>>(actS1, F(IX_WIN1), F(IX_BIN1), actS2);
  for (int blk = 2; blk < 4; ++blk) {
    int base = blkBase[blk];
    const int Mtiles = BB * TT2 / 16;         // 15988
    k_tds_conv<<<BB * ((TT2 + 63) / 64), 256, 0, stream>>>(
        actS2, F(base + 0), F(base + 1), F(base + 2), F(base + 3),
        y16, y32, 8, 8, 11, TT2);
    k_fc1_t<64><<<(Mtiles + 7) / 8, 256, 0, stream>>>(y16, w1f[blk], F(base + 5),
                                                      h16, Mtiles);
    k_fc2ln_t<64><<<(Mtiles + 7) / 8, 256, 0, stream>>>(h16, w2f[blk], F(base + 7),
                                                        y32, F(base + 8), F(base + 9),
                                                        actS2, Mtiles, TT2);
  }

  // ---- decoder ----
  k_interp_feats<<<(TTt * 64 * 64) / 256, 256, 0, stream>>>(actS2, ft16);
  k_lstm<<<1, 1024, 0, stream>>>(ft16, Wi0f, Wh0f, Wi1f, Wh1f, Wof,
                                 bz0, bz1, F(IX_BO), poses);
  k_interp_out<<<(BB * TT0 * DOUT) / 256, 256, 0, stream>>>(poses, (float*)d_out);
}
